// Llama4TextMoe_90031104459226
// MI455X (gfx1250) — matrix operand order repaired and checked by reading, not compiled, not run on hardware
//
#include <hip/hip_runtime.h>
#include <stdint.h>

#define TT   4096   // tokens = B*S
#define HH   2048   // hidden
#define II   1024   // routed intermediate
#define ISZ  4096   // shared intermediate
#define EE   16     // experts

#define BM   128    // block tile M
#define BN   64     // block tile N
#define BK   64     // block tile K

typedef __attribute__((ext_vector_type(16))) __bf16 v16bf;
typedef __attribute__((ext_vector_type(8)))  float  v8f;

union Frag16 { uint4 u[2]; v16bf f; };

// ---- fp32 -> bf16 (packed hardware converter when available) -----------
#if __has_builtin(__builtin_amdgcn_cvt_pk_bf16_f32)
__device__ __forceinline__ unsigned int f2bf_pk(float a, float b) {
  auto r = __builtin_amdgcn_cvt_pk_bf16_f32(a, b);
  union { decltype(r) v; unsigned int u; } cv;
  cv.v = r;
  return cv.u;
}
#else
__device__ __forceinline__ unsigned int f2bf_pk(float a, float b) {
  unsigned int ua = __float_as_uint(a), ub = __float_as_uint(b);
  ua += 0x7FFFu + ((ua >> 16) & 1u);
  ub += 0x7FFFu + ((ub >> 16) & 1u);
  return (ua >> 16) | (ub & 0xFFFF0000u);
}
#endif

__device__ __forceinline__ unsigned short f2bf1(float x) {
  return (unsigned short)f2bf_pk(x, 0.f);
}

__device__ __forceinline__ v8f wmma_bf16(const Frag16& a, const Frag16& b, v8f c) {
  return __builtin_amdgcn_wmma_f32_16x16x32_bf16(false, a.f, false, b.f,
                                                 (short)0, c, false, false);
}

// ---- tile loaders -------------------------------------------------------
// A tile: BM(128) x BK(64) bf16, row-major [128][64] in LDS. Source: bf16.
__device__ __forceinline__ void load_a128(unsigned short* As, const unsigned short* X,
                                          int ldx, int r0, int k0, int tid) {
  int r = tid >> 1, cg = (tid & 1) * 32;
  const unsigned short* src = X + (size_t)(r0 + r) * ldx + k0 + cg;
  unsigned short* dst = As + r * 64 + cg;
  uint4 v0 = ((const uint4*)src)[0];
  uint4 v1 = ((const uint4*)src)[1];
  uint4 v2 = ((const uint4*)src)[2];
  uint4 v3 = ((const uint4*)src)[3];
  ((uint4*)dst)[0] = v0; ((uint4*)dst)[1] = v1;
  ((uint4*)dst)[2] = v2; ((uint4*)dst)[3] = v3;
}

// Gathered A tile via token list (rows beyond count -> zero).
__device__ __forceinline__ void load_a128_gather(unsigned short* As, const unsigned short* X,
                                                 int ldx, const int* toks, int k0, int tid) {
  int r = tid >> 1, cg = (tid & 1) * 32;
  int tok = toks[r];
  uint4 v0 = make_uint4(0u,0u,0u,0u), v1 = v0, v2 = v0, v3 = v0;
  if (tok >= 0) {
    const unsigned short* src = X + (size_t)tok * ldx + k0 + cg;
    v0 = ((const uint4*)src)[0];
    v1 = ((const uint4*)src)[1];
    v2 = ((const uint4*)src)[2];
    v3 = ((const uint4*)src)[3];
  }
  unsigned short* dst = As + r * 64 + cg;
  ((uint4*)dst)[0] = v0; ((uint4*)dst)[1] = v1;
  ((uint4*)dst)[2] = v2; ((uint4*)dst)[3] = v3;
}

// B tile: BK(64) x BN(64) fp32 in global, converted to bf16, stored N-major
// [64 n][64 k] in LDS so a B fragment is two contiguous 16B LDS reads.
__device__ __forceinline__ void load_b64(unsigned short* Bt, const float* W,
                                         int ldw, int k0, int n0, int tid) {
  int k = tid >> 2, ng = (tid & 3) * 16;
  const float* p = W + (size_t)(k0 + k) * ldw + n0 + ng;
  float4 a = ((const float4*)p)[0];
  float4 b = ((const float4*)p)[1];
  float4 c = ((const float4*)p)[2];
  float4 d = ((const float4*)p)[3];
  unsigned short* q = Bt + ng * 64 + k;
  unsigned int u;
  u = f2bf_pk(a.x, a.y); q[ 0*64] = (unsigned short)u; q[ 1*64] = (unsigned short)(u >> 16);
  u = f2bf_pk(a.z, a.w); q[ 2*64] = (unsigned short)u; q[ 3*64] = (unsigned short)(u >> 16);
  u = f2bf_pk(b.x, b.y); q[ 4*64] = (unsigned short)u; q[ 5*64] = (unsigned short)(u >> 16);
  u = f2bf_pk(b.z, b.w); q[ 6*64] = (unsigned short)u; q[ 7*64] = (unsigned short)(u >> 16);
  u = f2bf_pk(c.x, c.y); q[ 8*64] = (unsigned short)u; q[ 9*64] = (unsigned short)(u >> 16);
  u = f2bf_pk(c.z, c.w); q[10*64] = (unsigned short)u; q[11*64] = (unsigned short)(u >> 16);
  u = f2bf_pk(d.x, d.y); q[12*64] = (unsigned short)u; q[13*64] = (unsigned short)(u >> 16);
  u = f2bf_pk(d.z, d.w); q[14*64] = (unsigned short)u; q[15*64] = (unsigned short)(u >> 16);
}

// ---- fragment loads (contiguous k = 16h + s in both A and B, ldk = 64) -----
__device__ __forceinline__ void frag_a(Frag16& f, const unsigned short* As, int kk, int lane) {
  int m = lane & 15, hi = (lane >> 4) * 16;
  f.u[0] = *(const uint4*)(As + m * 64 + kk + hi);
  f.u[1] = *(const uint4*)(As + m * 64 + kk + 8 + hi);
}

__device__ __forceinline__ void frag_b(Frag16& f, const unsigned short* Bt, int kk, int lane) {
  int n = lane & 15, hi = (lane >> 4) * 16;
  const uint4* p = (const uint4*)(Bt + n * 64 + kk + hi);
  f.u[0] = p[0]; f.u[1] = p[1];
}

__device__ __forceinline__ float silu_mul(float g, float u) {
  return u * (g / (1.f + __expf(-g)));
}

// ======================= kernels ========================================

__global__ void k_init(int* counts) {
  if (threadIdx.x < EE) counts[threadIdx.x] = 0;
}

// one block per token: 16 logits, top-1, sigmoid, append to expert list
__global__ __launch_bounds__(256) void k_router(const float* __restrict__ x,
                                                const float* __restrict__ rw,
                                                float* __restrict__ scores,
                                                float* __restrict__ scale,
                                                int* __restrict__ counts,
                                                int* __restrict__ tlist) {
  __shared__ float xs[HH];
  __shared__ float partial[256];
  int t = blockIdx.x, tid = threadIdx.x;
  for (int i = tid; i < HH; i += 256) xs[i] = x[(size_t)t * HH + i];
  __syncthreads();
  int e = tid >> 4, j = tid & 15;
  float acc = 0.f;
  for (int h = j; h < HH; h += 16) acc += xs[h] * rw[(size_t)e * HH + h];
  partial[tid] = acc;
  __syncthreads();
  for (int s = 8; s > 0; s >>= 1) {
    if (j < s) partial[tid] += partial[tid + s];
    __syncthreads();
  }
  if (tid == 0) {
    float best = partial[0]; int bi = 0;
    for (int k = 1; k < EE; k++) {
      float v = partial[k * 16];
      if (v > best) { best = v; bi = k; }
    }
    float s = 1.f / (1.f + __expf(-best));
    scale[t] = s;
    int pos = atomicAdd(&counts[bi], 1);
    tlist[bi * TT + pos] = t;
    for (int k = 0; k < EE; k++) scores[(size_t)k * TT + t] = (k == bi) ? s : 0.f;
  }
}

// x (fp32) -> xb (bf16 raw) and xsb (bf16 pre-scaled); fully vectorized
__global__ __launch_bounds__(256) void k_convert(const float* __restrict__ x,
                                                 const float* __restrict__ scale,
                                                 unsigned short* __restrict__ xb,
                                                 unsigned short* __restrict__ xsb) {
  int t = blockIdx.x, tid = threadIdx.x;
  float s = scale[t];
  const float4* xr = (const float4*)(x + (size_t)t * HH) + tid * 2;
  float4 a = xr[0], b = xr[1];
  uint4 raw, sc;
  raw.x = f2bf_pk(a.x, a.y);       raw.y = f2bf_pk(a.z, a.w);
  raw.z = f2bf_pk(b.x, b.y);       raw.w = f2bf_pk(b.z, b.w);
  sc.x  = f2bf_pk(a.x*s, a.y*s);   sc.y  = f2bf_pk(a.z*s, a.w*s);
  sc.z  = f2bf_pk(b.x*s, b.y*s);   sc.w  = f2bf_pk(b.z*s, b.w*s);
  *(uint4*)(xb  + (size_t)t * HH + tid * 8) = raw;
  *(uint4*)(xsb + (size_t)t * HH + tid * 8) = sc;
}

// --- dual-B compute stage: load all 6 fragments, then 8 back-to-back WMMAs
#define DUAL_STAGE(As_, B0_, B1_, acc0_, acc1_)                              \
  _Pragma("unroll")                                                          \
  for (int kk = 0; kk < BK; kk += 32) {                                      \
    Frag16 fa0, fa1, fb00, fb01, fb10, fb11;                                 \
    frag_a(fa0, (As_) + (wm * 32 +  0) * 64, kk, lane);                      \
    frag_a(fa1, (As_) + (wm * 32 + 16) * 64, kk, lane);                      \
    frag_b(fb00, (B0_) + (wn * 32 +  0) * 64, kk, lane);                     \
    frag_b(fb01, (B0_) + (wn * 32 + 16) * 64, kk, lane);                     \
    frag_b(fb10, (B1_) + (wn * 32 +  0) * 64, kk, lane);                     \
    frag_b(fb11, (B1_) + (wn * 32 + 16) * 64, kk, lane);                     \
    acc0_[0][0] = wmma_bf16(fa0, fb00, acc0_[0][0]);                         \
    acc0_[1][0] = wmma_bf16(fa1, fb00, acc0_[1][0]);                         \
    acc0_[0][1] = wmma_bf16(fa0, fb01, acc0_[0][1]);                         \
    acc0_[1][1] = wmma_bf16(fa1, fb01, acc0_[1][1]);                         \
    acc1_[0][0] = wmma_bf16(fa0, fb10, acc1_[0][0]);                         \
    acc1_[1][0] = wmma_bf16(fa1, fb10, acc1_[1][0]);                         \
    acc1_[0][1] = wmma_bf16(fa0, fb11, acc1_[0][1]);                         \
    acc1_[1][1] = wmma_bf16(fa1, fb11, acc1_[1][1]);                         \
  }

// --- single-B compute stage: load 4 fragments, then 4 back-to-back WMMAs
#define SINGLE_STAGE(As_, B0_, acc_)                                         \
  _Pragma("unroll")                                                          \
  for (int kk = 0; kk < BK; kk += 32) {                                      \
    Frag16 fa0, fa1, fb0, fb1;                                               \
    frag_a(fa0, (As_) + (wm * 32 +  0) * 64, kk, lane);                      \
    frag_a(fa1, (As_) + (wm * 32 + 16) * 64, kk, lane);                      \
    frag_b(fb0, (B0_) + (wn * 32 +  0) * 64, kk, lane);                      \
    frag_b(fb1, (B0_) + (wn * 32 + 16) * 64, kk, lane);                      \
    acc_[0][0] = wmma_bf16(fa0, fb0, acc_[0][0]);                            \
    acc_[1][0] = wmma_bf16(fa1, fb0, acc_[1][0]);                            \
    acc_[0][1] = wmma_bf16(fa0, fb1, acc_[0][1]);                            \
    acc_[1][1] = wmma_bf16(fa1, fb1, acc_[1][1]);                            \
  }

// shared path GEMM1: hs = silu(x@Wg) * (x@Wu), bf16 out. 128x64x64 tiles.
__global__ __launch_bounds__(256) void k_shared_gateup(const unsigned short* __restrict__ xb,
                                                       const float* __restrict__ Wg,
                                                       const float* __restrict__ Wu,
                                                       unsigned short* __restrict__ hs) {
  __shared__ __align__(16) unsigned short As[BM * BK];
  __shared__ __align__(16) unsigned short Bg[BN * BK];
  __shared__ __align__(16) unsigned short Bu[BN * BK];
  int n0 = blockIdx.x * BN, t0 = blockIdx.y * BM;
  int tid = threadIdx.x, lane = tid & 31, w = tid >> 5;
  int wm = w & 3, wn = w >> 2;
  v8f ag[2][2] = {}, au[2][2] = {};
  for (int k0 = 0; k0 < HH; k0 += BK) {
    load_a128(As, xb, HH, t0, k0, tid);
    load_b64(Bg, Wg, ISZ, k0, n0, tid);
    load_b64(Bu, Wu, ISZ, k0, n0, tid);
    if (k0 + BK < HH) {
      __builtin_prefetch(Wg + (size_t)(k0 + BK + (tid >> 2)) * ISZ + n0 + (tid & 3) * 16, 0, 0);
      __builtin_prefetch(Wu + (size_t)(k0 + BK + (tid >> 2)) * ISZ + n0 + (tid & 3) * 16, 0, 0);
    }
    __syncthreads();
    DUAL_STAGE(As, Bg, Bu, ag, au)
    __syncthreads();
  }
  int nl = lane & 15, mo = (lane >> 4) * 8;
#pragma unroll
  for (int mi = 0; mi < 2; mi++) {
#pragma unroll
    for (int r = 0; r < 8; r++) {
      int row = t0 + wm * 32 + mi * 16 + mo + r;
      size_t base = (size_t)row * ISZ + n0 + wn * 32 + nl;
      hs[base]      = f2bf1(silu_mul(ag[mi][0][r], au[mi][0][r]));
      hs[base + 16] = f2bf1(silu_mul(ag[mi][1][r], au[mi][1][r]));
    }
  }
}

// shared path GEMM2: out = hs @ Wd (plain store; routed path adds later)
__global__ __launch_bounds__(256) void k_shared_down(const unsigned short* __restrict__ hs,
                                                     const float* __restrict__ Wd,
                                                     float* __restrict__ out) {
  __shared__ __align__(16) unsigned short As[BM * BK];
  __shared__ __align__(16) unsigned short Bt[BN * BK];
  int n0 = blockIdx.x * BN, t0 = blockIdx.y * BM;
  int tid = threadIdx.x, lane = tid & 31, w = tid >> 5;
  int wm = w & 3, wn = w >> 2;
  v8f acc[2][2] = {};
  for (int k0 = 0; k0 < ISZ; k0 += BK) {
    load_a128(As, hs, ISZ, t0, k0, tid);
    load_b64(Bt, Wd, HH, k0, n0, tid);
    if (k0 + BK < ISZ)
      __builtin_prefetch(Wd + (size_t)(k0 + BK + (tid >> 2)) * HH + n0 + (tid & 3) * 16, 0, 0);
    __syncthreads();
    SINGLE_STAGE(As, Bt, acc)
    __syncthreads();
  }
  int nl = lane & 15, mo = (lane >> 4) * 8;
#pragma unroll
  for (int mi = 0; mi < 2; mi++) {
#pragma unroll
    for (int r = 0; r < 8; r++) {
      int row = t0 + wm * 32 + mi * 16 + mo + r;
      size_t base = (size_t)row * HH + n0 + wn * 32 + nl;
      out[base]      = acc[mi][0][r];
      out[base + 16] = acc[mi][1][r];
    }
  }
}

// routed gate+up for one expert's gathered tokens -> act (bf16, token rows)
__global__ __launch_bounds__(256) void k_routed_gateup(const unsigned short* __restrict__ xsb,
                                                       const float* __restrict__ gup,
                                                       const int* __restrict__ counts,
                                                       const int* __restrict__ tlist,
                                                       unsigned short* __restrict__ act) {
  int e = blockIdx.z;
  int cnt = counts[e];
  int m0 = blockIdx.y * BM;
  if (m0 >= cnt) return;
  __shared__ int toks[BM];
  __shared__ __align__(16) unsigned short As[BM * BK];
  __shared__ __align__(16) unsigned short Bg[BN * BK];
  __shared__ __align__(16) unsigned short Bu[BN * BK];
  int tid = threadIdx.x, lane = tid & 31, w = tid >> 5;
  int wm = w & 3, wn = w >> 2;
  if (tid < BM) toks[tid] = (m0 + tid < cnt) ? tlist[e * TT + m0 + tid] : -1;
  __syncthreads();
  const float* We = gup + (size_t)e * HH * (2 * II);
  int n0 = blockIdx.x * BN;
  v8f ag[2][2] = {}, au[2][2] = {};
  for (int k0 = 0; k0 < HH; k0 += BK) {
    load_a128_gather(As, xsb, HH, toks, k0, tid);
    load_b64(Bg, We, 2 * II, k0, n0, tid);
    load_b64(Bu, We, 2 * II, k0, II + n0, tid);
    __syncthreads();
    DUAL_STAGE(As, Bg, Bu, ag, au)
    __syncthreads();
  }
  int nl = lane & 15, mo = (lane >> 4) * 8;
#pragma unroll
  for (int mi = 0; mi < 2; mi++) {
#pragma unroll
    for (int r = 0; r < 8; r++) {
      int tok = toks[wm * 32 + mi * 16 + mo + r];
      if (tok < 0) continue;
      size_t base = (size_t)tok * II + n0 + wn * 32 + nl;
      act[base]      = f2bf1(silu_mul(ag[mi][0][r], au[mi][0][r]));
      act[base + 16] = f2bf1(silu_mul(ag[mi][1][r], au[mi][1][r]));
    }
  }
}

// routed down-proj: out[token] += act[token] @ down[e]  (token rows unique)
__global__ __launch_bounds__(256) void k_routed_down(const unsigned short* __restrict__ act,
                                                     const float* __restrict__ dwn,
                                                     const int* __restrict__ counts,
                                                     const int* __restrict__ tlist,
                                                     float* __restrict__ out) {
  int e = blockIdx.z;
  int cnt = counts[e];
  int m0 = blockIdx.y * BM;
  if (m0 >= cnt) return;
  __shared__ int toks[BM];
  __shared__ __align__(16) unsigned short As[BM * BK];
  __shared__ __align__(16) unsigned short Bt[BN * BK];
  int tid = threadIdx.x, lane = tid & 31, w = tid >> 5;
  int wm = w & 3, wn = w >> 2;
  if (tid < BM) toks[tid] = (m0 + tid < cnt) ? tlist[e * TT + m0 + tid] : -1;
  __syncthreads();
  const float* We = dwn + (size_t)e * II * HH;
  int n0 = blockIdx.x * BN;
  v8f acc[2][2] = {};
  for (int k0 = 0; k0 < II; k0 += BK) {
    load_a128_gather(As, act, II, toks, k0, tid);
    load_b64(Bt, We, HH, k0, n0, tid);
    __syncthreads();
    SINGLE_STAGE(As, Bt, acc)
    __syncthreads();
  }
  int nl = lane & 15, mo = (lane >> 4) * 8;
#pragma unroll
  for (int mi = 0; mi < 2; mi++) {
#pragma unroll
    for (int r = 0; r < 8; r++) {
      int tok = toks[wm * 32 + mi * 16 + mo + r];
      if (tok < 0) continue;
      size_t base = (size_t)tok * HH + n0 + wn * 32 + nl;
      out[base]      += acc[mi][0][r];
      out[base + 16] += acc[mi][1][r];
    }
  }
}

// ======================= launch =========================================
extern "C" void kernel_launch(void* const* d_in, const int* in_sizes, int n_in,
                              void* d_out, int out_size, void* d_ws, size_t ws_size,
                              hipStream_t stream) {
  (void)in_sizes; (void)n_in; (void)out_size; (void)ws_size;
  const float* x   = (const float*)d_in[0];   // (T, H)
  const float* rw  = (const float*)d_in[1];   // (E, H)
  const float* gup = (const float*)d_in[2];   // (E, H, 2I)
  const float* dwn = (const float*)d_in[3];   // (E, I, H)
  const float* wg  = (const float*)d_in[4];   // (H, IS)
  const float* wu  = (const float*)d_in[5];   // (H, IS)
  const float* wd  = (const float*)d_in[6];   // (IS, H)

  float* out    = (float*)d_out;              // (T, H)
  float* scores = out + (size_t)TT * HH;      // (E, T)

  char* w = (char*)d_ws;
  int*   counts = (int*)w;
  float* scale  = (float*)(w + 256);
  int*   tlist  = (int*)(w + 256 + TT * 4);
  size_t off = 256 + (size_t)TT * 4 + (size_t)EE * TT * 4;
  unsigned short* xb  = (unsigned short*)(w + off); off += (size_t)TT * HH * 2;
  unsigned short* xsb = (unsigned short*)(w + off); off += (size_t)TT * HH * 2;
  unsigned short* hs  = (unsigned short*)(w + off); off += (size_t)TT * ISZ * 2;
  unsigned short* act = (unsigned short*)(w + off);

  k_init<<<1, 64, 0, stream>>>(counts);
  k_router<<<TT, 256, 0, stream>>>(x, rw, scores, scale, counts, tlist);
  k_convert<<<TT, 256, 0, stream>>>(x, scale, xb, xsb);
  k_shared_gateup<<<dim3(ISZ / BN, TT / BM), 256, 0, stream>>>(xb, wg, wu, hs);
  k_shared_down<<<dim3(HH / BN, TT / BM), 256, 0, stream>>>(hs, wd, out);
  k_routed_gateup<<<dim3(II / BN, TT / BM, EE), 256, 0, stream>>>(xsb, gup, counts, tlist, act);
  k_routed_down<<<dim3(HH / BN, TT / BM, EE), 256, 0, stream>>>(act, dwn, counts, tlist, out);
}
